// LGCN_mlp2_9706626089564
// MI455X (gfx1250) — compile-verified, matrix operand order checked by reading, not run on hardware
//
#include <hip/hip_runtime.h>
#include <stdint.h>

#define NN    50000
#define FEATN 128
#define EE    512000
#define KHOP  8
#define HID   512
#define OUTC  32
#define KCAT  ((KHOP + 1) * FEATN)   /* 1152 */
#define EPSF  1e-5f

typedef __attribute__((ext_vector_type(16))) __bf16          v16bf;
typedef __attribute__((ext_vector_type(8)))  float           v8f;
typedef __attribute__((ext_vector_type(8)))  unsigned short  us8;

__device__ __forceinline__ unsigned short f2bf(float f) {
  unsigned int u = __float_as_uint(f);
  u += 0x7FFFu + ((u >> 16) & 1u);     // round-to-nearest-even
  return (unsigned short)(u >> 16);
}

// ---- fragment load: per-lane two contiguous 8-elem K-runs (ISA 7.12.2 16-bit A layout)
__device__ __forceinline__ v16bf load_frag(const unsigned short* base, int half) {
  union U { struct { us8 lo, hi; } s; v16bf v; } u;
  u.s.lo = *(const us8*)(base + half * 8);        // K = half*8 .. +7
  u.s.hi = *(const us8*)(base + 16 + half * 8);   // K = 16+half*8 .. +7
  return u.v;
}

// ---- CDNA5 async memory->LDS (no register relay, tracked by ASYNCcnt).
// LDS VDST operand takes the zero-based LDS offset = low 32 bits of the flat
// LDS-aperture address (ISA 10.2).
__device__ __forceinline__ void async_ld_b128(unsigned short* lds_dst, const unsigned short* gsrc) {
  unsigned loff = (unsigned)(uintptr_t)lds_dst;
  unsigned long long ga = (unsigned long long)(uintptr_t)gsrc;
  asm volatile("global_load_async_to_lds_b128 %0, %1, off" :: "v"(loff), "v"(ga) : "memory");
}
__device__ __forceinline__ void wait_async0() {
  asm volatile("s_wait_asynccnt 0" ::: "memory");
}

__device__ __forceinline__ v8f wmma_bf16(v16bf a, v16bf b, v8f c) {
  return __builtin_amdgcn_wmma_f32_16x16x32_bf16(false, a, false, b, (short)0, c, false, false);
}

// =======================  small utility kernels  =======================
__global__ void zero_kernel(float* p, size_t n) {
  size_t i = (size_t)blockIdx.x * blockDim.x + threadIdx.x;
  if (i < n) p[i] = 0.f;
}

__global__ void edge_w_kernel(const float* __restrict__ coor,
                              const int* __restrict__ src, const int* __restrict__ dst,
                              float* __restrict__ w, float* __restrict__ deg) {
  int e = blockIdx.x * blockDim.x + threadIdx.x;
  if (e >= EE) return;
  int s = src[e], d = dst[e];
  float dx = coor[2 * s]     - coor[2 * d];
  float dy = coor[2 * s + 1] - coor[2 * d + 1];
  float wv = __expf(-(dx * dx + dy * dy));
  w[e] = wv;
  atomicAdd(&deg[d], wv);
}

__global__ void edge_norm_kernel(const float* __restrict__ w, const float* __restrict__ deg,
                                 const int* __restrict__ dst, float* __restrict__ wn) {
  int e = blockIdx.x * blockDim.x + threadIdx.x;
  if (e >= EE) return;
  wn[e] = w[e] / (deg[dst[e]] + EPSF);
}

// one wave per edge: 32 lanes x float4 = 128 features; x (25.6MB) is L2-resident
__global__ __launch_bounds__(256) void prop_kernel(const float* __restrict__ x,
                                                   const float* __restrict__ wn,
                                                   const int* __restrict__ src,
                                                   const int* __restrict__ dst,
                                                   float* __restrict__ xn) {
  int gt   = blockIdx.x * blockDim.x + threadIdx.x;
  int e    = gt >> 5;
  int lane = gt & 31;
  if (e >= EE) return;
  int s = src[e], d = dst[e];
  float wv = wn[e];
  const float4 xv = *(const float4*)(x + (size_t)s * FEATN + lane * 4);
  float* o = xn + (size_t)d * FEATN + lane * 4;
  atomicAdd(o + 0, wv * xv.x);
  atomicAdd(o + 1, wv * xv.y);
  atomicAdd(o + 2, wv * xv.z);
  atomicAdd(o + 3, wv * xv.w);
}

// copy hop output into bf16 Xcat block `hop`, optionally zero the next ping-pong buffer
__global__ void cvt_hop_kernel(const float* __restrict__ xsrc, unsigned short* __restrict__ Xcat,
                               float* __restrict__ xzero, int hop) {
  int i = blockIdx.x * blockDim.x + threadIdx.x;
  if (i >= NN * FEATN) return;
  int n = i >> 7, f = i & 127;
  Xcat[(size_t)n * KCAT + hop * FEATN + f] = f2bf(xsrc[i]);
  if (xzero) xzero[i] = 0.f;
}

// weights -> bf16, K-chunk swizzled: Bsw[(k/32)][n][k%32]  (16B-granular staging)
__global__ void w1t_kernel(const float* __restrict__ W1, unsigned short* __restrict__ Bsw) {
  int i = blockIdx.x * blockDim.x + threadIdx.x;
  if (i >= HID * KCAT) return;
  int h = i / KCAT, k = i % KCAT;
  Bsw[(size_t)(k >> 5) * (HID * 32) + h * 32 + (k & 31)] = f2bf(W1[i]);
}
__global__ void w2t_kernel(const float* __restrict__ W2, unsigned short* __restrict__ Bsw) {
  int i = blockIdx.x * blockDim.x + threadIdx.x;
  if (i >= OUTC * HID) return;
  int o = i / HID, k = i % HID;
  Bsw[(size_t)(k >> 5) * (OUTC * 32) + o * 32 + (k & 31)] = f2bf(W2[i]);
}

// =======================  main GEMM: 128x64 block tile, async double-buffered  =======================
// C = A[M x K] * B + bias (+leaky). B given swizzled: Bsw[k/32][nb][32].
// 8 waves in 4x2 grid; each wave: 32x32 register tile = 4 WMMAs per 32-K step.
__global__ __launch_bounds__(256) void gemm_bf16_128x64_kernel(
    const unsigned short* __restrict__ A, int lda,
    const unsigned short* __restrict__ Bsw, int nb,
    const float* __restrict__ bias,
    float* __restrict__ C, int ldc,
    int M, int K, int leaky) {
  __shared__ __align__(16) unsigned short sA[2][128 * 40];  // 2 x 10 KB
  __shared__ __align__(16) unsigned short sB[2][64 * 40];   // 2 x 5 KB

  const int tid  = threadIdx.x;
  const int wave = tid >> 5;
  const int lane = tid & 31;
  const int m0 = blockIdx.x * 128;
  const int n0 = blockIdx.y * 64;
  const int wr = wave >> 1;            // 0..3 : 32-row group
  const int wc = wave & 1;             // 0..1 : 32-col group
  const int lrow = lane & 15;
  const int half = lane >> 4;

  v8f acc00 = {}, acc01 = {}, acc10 = {}, acc11 = {};

  // A staging: 512 16B chunks / 256 threads -> 2 each (rows tid>>2 and 64+tid>>2)
  const int ar0 = tid >> 2, aseg = tid & 3;
  const int ar1 = ar0 + 64;
  int gm0 = m0 + ar0; if (gm0 >= M) gm0 = M - 1;
  int gm1 = m0 + ar1; if (gm1 >= M) gm1 = M - 1;
  const unsigned short* ap0 = A + (size_t)gm0 * lda + aseg * 8;
  const unsigned short* ap1 = A + (size_t)gm1 * lda + aseg * 8;
  // B staging: 256 16B chunks -> 1 each (col tid>>2, seg tid&3), fully coalesced
  const int bcol = tid >> 2, bseg = tid & 3;
  const unsigned short* bp = Bsw + (size_t)(n0 + bcol) * 32 + bseg * 8;
  const size_t bstep = (size_t)nb * 32;        // bytes-in-elems per 32-K chunk

  auto issue = [&](int stg, int k0) {
    async_ld_b128(&sA[stg][ar0 * 40 + aseg * 8], ap0 + k0);
    async_ld_b128(&sA[stg][ar1 * 40 + aseg * 8], ap1 + k0);
    async_ld_b128(&sB[stg][bcol * 40 + bseg * 8], bp + (size_t)(k0 >> 5) * bstep);
  };

  issue(0, 0);
  int stg = 0;
  for (int k0 = 0; k0 < K; k0 += 32) {
    wait_async0();        // this wave's stage-stg loads landed in LDS
    __syncthreads();      // everyone's landed; prior stage reads all retired
    if (k0 + 32 < K) issue(stg ^ 1, k0 + 32);

    v16bf fa0 = load_frag(&sA[stg][(wr * 32 + lrow) * 40], half);
    v16bf fa1 = load_frag(&sA[stg][(wr * 32 + 16 + lrow) * 40], half);
    v16bf fb0 = load_frag(&sB[stg][(wc * 32 + lrow) * 40], half);
    v16bf fb1 = load_frag(&sB[stg][(wc * 32 + 16 + lrow) * 40], half);
    acc00 = wmma_bf16(fa0, fb0, acc00);
    acc01 = wmma_bf16(fa0, fb1, acc01);
    acc10 = wmma_bf16(fa1, fb0, acc10);
    acc11 = wmma_bf16(fa1, fb1, acc11);
    stg ^= 1;
  }

  // D layout: lane -> col lrow, rows half*8 + j
  const int ncol0 = n0 + wc * 32 + lrow;
  const int mrow0 = m0 + wr * 32 + half * 8;
  const float bv0 = bias[ncol0];
  const float bv1 = bias[ncol0 + 16];
#pragma unroll
  for (int j = 0; j < 8; ++j) {
    int mA = mrow0 + j;
    int mB = mrow0 + 16 + j;
    if (mA < M) {
      float v0 = acc00[j] + bv0, v1 = acc01[j] + bv1;
      if (leaky) { v0 = (v0 >= 0.f) ? v0 : 0.2f * v0; v1 = (v1 >= 0.f) ? v1 : 0.2f * v1; }
      C[(size_t)mA * ldc + ncol0]      = v0;
      C[(size_t)mA * ldc + ncol0 + 16] = v1;
    }
    if (mB < M) {
      float v0 = acc10[j] + bv0, v1 = acc11[j] + bv1;
      if (leaky) { v0 = (v0 >= 0.f) ? v0 : 0.2f * v0; v1 = (v1 >= 0.f) ? v1 : 0.2f * v1; }
      C[(size_t)mB * ldc + ncol0]      = v0;
      C[(size_t)mB * ldc + ncol0 + 16] = v1;
    }
  }
}

// =======================  small GEMM (N=32): 64x32 tile, 1 WMMA tile per wave  =======================
__global__ __launch_bounds__(256) void gemm_bf16_64x32_kernel(
    const unsigned short* __restrict__ A, int lda,
    const unsigned short* __restrict__ Bsw, int nb,
    const float* __restrict__ bias,
    float* __restrict__ C, int ldc,
    int M, int K, int leaky) {
  __shared__ __align__(16) unsigned short sA[64 * 40];
  __shared__ __align__(16) unsigned short sB[32 * 40];

  const int tid  = threadIdx.x;
  const int wave = tid >> 5;
  const int lane = tid & 31;
  const int m0 = blockIdx.x * 64;
  const int n0 = blockIdx.y * 32;
  const int rt = wave >> 1;
  const int ct = wave & 1;
  const int lrow = lane & 15;
  const int half = lane >> 4;

  v8f acc = {};

  const int arow = tid >> 2, aseg = tid & 3;
  int gm = m0 + arow; if (gm >= M) gm = M - 1;
  const unsigned short* aptr = A + (size_t)gm * lda + aseg * 8;
  const int bcol = (tid & 127) >> 2, bseg = tid & 3;
  const unsigned short* bp = Bsw + (size_t)(n0 + bcol) * 32 + bseg * 8;
  const size_t bstep = (size_t)nb * 32;

  for (int k0 = 0; k0 < K; k0 += 32) {
    *(us8*)(sA + arow * 40 + aseg * 8) = *(const us8*)(aptr + k0);
    if (tid < 128)
      *(us8*)(sB + bcol * 40 + bseg * 8) = *(const us8*)(bp + (size_t)(k0 >> 5) * bstep);

    __syncthreads();
    v16bf af  = load_frag(sA + (rt * 16 + lrow) * 40, half);
    v16bf bfv = load_frag(sB + (ct * 16 + lrow) * 40, half);
    acc = wmma_bf16(af, bfv, acc);
    __syncthreads();
  }

  const int n = n0 + ct * 16 + lrow;
  const int mbase = m0 + rt * 16 + half * 8;
  const float bv = bias[n];
#pragma unroll
  for (int j = 0; j < 8; ++j) {
    int m = mbase + j;
    if (m < M) {
      float v = acc[j] + bv;
      if (leaky) v = (v >= 0.f) ? v : 0.2f * v;
      C[(size_t)m * ldc + n] = v;
    }
  }
}

// =======================  batch-norm  =======================
__global__ __launch_bounds__(256) void bn_stats_kernel(const float* __restrict__ H,
                                                       float* __restrict__ sums) {
  int r0 = blockIdx.x * 128;
  int rend = r0 + 128; if (rend > NN) rend = NN;
  int tid = threadIdx.x;
#pragma unroll
  for (int cc = 0; cc < 2; ++cc) {
    int c = tid + cc * 256;
    float s = 0.f, s2 = 0.f;
    for (int r = r0; r < rend; ++r) {
      float v = H[(size_t)r * HID + c];
      s += v; s2 += v * v;
    }
    atomicAdd(&sums[c], s);
    atomicAdd(&sums[HID + c], s2);
  }
}

__global__ void bn_norm_kernel(const float* __restrict__ H, const float* __restrict__ sums,
                               const float* __restrict__ gamma, const float* __restrict__ beta,
                               unsigned short* __restrict__ Hn) {
  size_t i = (size_t)blockIdx.x * blockDim.x + threadIdx.x;
  if (i >= (size_t)NN * HID) return;
  int c = (int)(i & (HID - 1));
  const float invM = 1.f / (float)NN;
  float mu  = sums[c] * invM;
  float var = sums[HID + c] * invM - mu * mu;
  float v = (H[i] - mu) * rsqrtf(var + EPSF) * gamma[c] + beta[c];
  Hn[i] = f2bf(v);
}

// =======================  host driver  =======================
extern "C" void kernel_launch(void* const* d_in, const int* in_sizes, int n_in,
                              void* d_out, int out_size, void* d_ws, size_t ws_size,
                              hipStream_t stream) {
  const float* feature = (const float*)d_in[0];
  const float* coor    = (const float*)d_in[1];
  const float* W1      = (const float*)d_in[2];
  const float* b1      = (const float*)d_in[3];
  const float* gamma   = (const float*)d_in[4];
  const float* beta    = (const float*)d_in[5];
  const float* W2      = (const float*)d_in[6];
  const float* b2      = (const float*)d_in[7];
  const int*   ei      = (const int*)d_in[8];
  const int* src = ei;
  const int* dst = ei + EE;
  (void)in_sizes; (void)n_in; (void)out_size; (void)ws_size;

  char* wsb = (char*)d_ws;
  size_t off = 0;
  auto take = [&](size_t bytes) -> char* {
    char* p = wsb + off;
    off = (off + bytes + 255) & ~(size_t)255;
    return p;
  };
  float* w    = (float*)take((size_t)EE * 4);
  float* wn   = (float*)take((size_t)EE * 4);
  float* deg  = (float*)take((size_t)NN * 4);
  float* xA   = (float*)take((size_t)NN * FEATN * 4);
  float* xB   = (float*)take((size_t)NN * FEATN * 4);
  unsigned short* Xcat = (unsigned short*)take((size_t)NN * KCAT * 2);
  float* H    = (float*)take((size_t)NN * HID * 4);
  unsigned short* Hn   = (unsigned short*)take((size_t)NN * HID * 2);
  unsigned short* W1t  = (unsigned short*)take((size_t)KCAT * HID * 2);
  unsigned short* W2t  = (unsigned short*)take((size_t)HID * OUTC * 2);
  float* sums = (float*)take(2 * HID * 4);

  // ---- edge weights + normalization
  zero_kernel<<<(NN + 255) / 256, 256, 0, stream>>>(deg, (size_t)NN);
  zero_kernel<<<((size_t)NN * FEATN + 255) / 256, 256, 0, stream>>>(xB, (size_t)NN * FEATN);
  edge_w_kernel<<<(EE + 255) / 256, 256, 0, stream>>>(coor, src, dst, w, deg);
  edge_norm_kernel<<<(EE + 255) / 256, 256, 0, stream>>>(w, deg, dst, wn);

  // ---- weight conversion (fp32 -> bf16, K-chunk swizzled)
  w1t_kernel<<<((HID * KCAT) + 255) / 256, 256, 0, stream>>>(W1, W1t);
  w2t_kernel<<<((OUTC * HID) + 255) / 256, 256, 0, stream>>>(W2, W2t);

  // ---- Xcat hop 0 = feature
  cvt_hop_kernel<<<((NN * FEATN) + 255) / 256, 256, 0, stream>>>(feature, Xcat, nullptr, 0);

  // ---- K hops of weighted message passing (ping-pong xA/xB)
  const int prop_blocks = ((EE * 32) + 255) / 256;
  const int cvt_blocks  = ((NN * FEATN) + 255) / 256;
  for (int k = 1; k <= KHOP; ++k) {
    const float* xin = (k == 1) ? feature : ((k & 1) ? xA : xB);
    float* xo = (k & 1) ? xB : xA;
    float* zt = (k < KHOP) ? ((k & 1) ? xA : xB) : nullptr;
    prop_kernel<<<prop_blocks, 256, 0, stream>>>(xin, wn, src, dst, xo);
    cvt_hop_kernel<<<cvt_blocks, 256, 0, stream>>>(xo, Xcat, zt, k);
  }

  // ---- GEMM1: H = leaky_relu(Xcat @ W1^T + b1)   [50000 x 1152] x [1152 x 512]
  {
    dim3 grid((NN + 127) / 128, HID / 64);
    gemm_bf16_128x64_kernel<<<grid, 256, 0, stream>>>(Xcat, KCAT, W1t, HID, b1, H, HID,
                                                      NN, KCAT, /*leaky=*/1);
  }

  // ---- BatchNorm (batch stats) -> bf16 Hn
  zero_kernel<<<(2 * HID + 255) / 256, 256, 0, stream>>>(sums, (size_t)(2 * HID));
  bn_stats_kernel<<<(NN + 127) / 128, 256, 0, stream>>>(H, sums);
  bn_norm_kernel<<<(((size_t)NN * HID) + 255) / 256, 256, 0, stream>>>(H, sums, gamma, beta, Hn);

  // ---- GEMM2: out = Hn @ W2^T + b2   [50000 x 512] x [512 x 32]
  {
    dim3 grid((NN + 63) / 64, OUTC / 32);
    gemm_bf16_64x32_kernel<<<grid, 256, 0, stream>>>(Hn, HID, W2t, OUTC, b2, (float*)d_out, OUTC,
                                                     NN, HID, /*leaky=*/0);
  }
}
